// PNALayer_69200513073222
// MI455X (gfx1250) — matrix operand order repaired and checked by reading, not compiled, not run on hardware
//
#include <hip/hip_runtime.h>
#include <math.h>

// PNA layer for MI455X (gfx1250, wave32, WMMA).
// Pipeline: swizzle weights -> init agg -> edge GEMM(bf16 WMMA)+atomic scatter
//           -> posttrans GEMM (mean/std finalize + scalers fused into A build)
//           -> mix GEMM + leaky-relu + residual.
// Agg buffers (~102MB) live in d_ws and fit MI455X's 192MB L2 for atomics.

#define DD 128
#define NTILE 8            // 128 / 16 output-column tiles
#define EPSV 1e-5f
#define AVGDLOG 1.6094379124341003f  // log(4+1)

typedef __attribute__((ext_vector_type(16))) __bf16 v16bf;
typedef __attribute__((ext_vector_type(8)))  float  v8f;

union Frag { unsigned u[8]; v16bf v; };

__device__ __forceinline__ unsigned bf16pack(float a, float b) {
  unsigned ua = __builtin_bit_cast(unsigned, a);
  unsigned ub = __builtin_bit_cast(unsigned, b);
  ua += 0x7FFFu + ((ua >> 16) & 1u);   // RNE
  ub += 0x7FFFu + ((ub >> 16) & 1u);
  return (ua >> 16) | (ub & 0xFFFF0000u);
}

// 16-bit A-matrix 16x32 layout (ISA 7.12.2): lane<16 holds K 0..7,16..23;
// lane>=16 holds K 8..15,24..31; VGPR v covers a consecutive K pair.
__device__ __forceinline__ int a_koff(int v, int lane) {
  return ((v >> 2) << 4) | ((v & 3) << 1) | (((lane >> 4) & 1) << 3);
}

__device__ __forceinline__ void load_bfrag(Frag& b, const unsigned* __restrict__ wswz,
                                           int ks, int nt, int lane) {
  const uint4* q = (const uint4*)(wswz + ((size_t)(ks * NTILE + nt) * 32 + lane) * 8);
  uint4 q0 = q[0], q1 = q[1];
  b.u[0] = q0.x; b.u[1] = q0.y; b.u[2] = q0.z; b.u[3] = q0.w;
  b.u[4] = q1.x; b.u[5] = q1.y; b.u[6] = q1.z; b.u[7] = q1.w;
}

// ---- weight pre-swizzle into B-fragment layout -----------------------------
// read index in hot loop: ((ks*NTILE + nt)*32 + lane)*8 + v
__global__ void swz_kernel(const float* __restrict__ W, unsigned* __restrict__ out,
                           int ksteps) {
  int idx = blockIdx.x * blockDim.x + threadIdx.x;
  int total = ksteps * NTILE * 32 * 8;
  if (idx >= total) return;
  int v    = idx & 7;
  int lane = (idx >> 3) & 31;
  int nt   = (idx >> 8) & 7;
  int ks   = idx >> 11;
  int n = nt * 16 + (lane & 15);
  int k = ks * 32 + a_koff(v, lane);   // B 32x16: lane=col, same K order as A (lanes 16..31 at K+8)
  out[idx] = bf16pack(W[(size_t)k * DD + n], W[(size_t)(k + 1) * DD + n]);
}

// ---- init aggregation buffers ----------------------------------------------
__global__ void init_kernel(float* sum, float* sumsq, int* mxb, int* mnb,
                            float* deg, int n_nodes) {
  int i = blockIdx.x * blockDim.x + threadIdx.x;
  int total = n_nodes * DD;
  if (i < total) {
    sum[i] = 0.0f; sumsq[i] = 0.0f;
    mxb[i] = 0;                 // e >= 0, so 0 is identity for max
    mnb[i] = 0x7F800000;        // +inf bits
  }
  if (i < n_nodes) deg[i] = 0.0f;
}

// ---- edge kernel: pretrans GEMM + relu + atomic segment scatter ------------
__global__ __launch_bounds__(256) void edge_kernel(
    const float* __restrict__ h, const int* __restrict__ esrc,
    const int* __restrict__ edst, const unsigned* __restrict__ wswz,
    const float* __restrict__ bias,
    float* __restrict__ sum, float* __restrict__ sumsq,
    int* __restrict__ mxb, int* __restrict__ mnb,
    float* __restrict__ deg, int n_edges) {
  const int lane = threadIdx.x & 31;
  const int wt = blockIdx.x * (blockDim.x >> 5) + (threadIdx.x >> 5);
  const int e0 = wt << 4;
  if (e0 >= n_edges) return;

  const int r = lane & 15;
  const int er = e0 + r;
  const bool rvalid = er < n_edges;
  const int src = rvalid ? esrc[er] : 0;
  const int dst = rvalid ? edst[er] : 0;
  const float* hs = h + (size_t)src * DD;
  const float* hd = h + (size_t)dst * DD;

  v8f acc[NTILE];
  #pragma unroll
  for (int nt = 0; nt < NTILE; ++nt) {
    float b = bias[nt * 16 + r];
    #pragma unroll
    for (int j = 0; j < 8; ++j) acc[nt][j] = b;
  }

  // K = 256: first 128 from h[src], last 128 from h[dst]
  #pragma unroll
  for (int ks = 0; ks < 8; ++ks) {
    const float* base = (ks < 4) ? hs : hd;
    const int kb = (ks & 3) * 32;
    Frag a;
    #pragma unroll
    for (int v = 0; v < 8; ++v) {
      const float2 p = *(const float2*)(base + kb + a_koff(v, lane));
      a.u[v] = bf16pack(p.x, p.y);
    }
    #pragma unroll
    for (int nt = 0; nt < NTILE; ++nt) {
      Frag b;
      load_bfrag(b, wswz, ks, nt, lane);
      acc[nt] = __builtin_amdgcn_wmma_f32_16x16x32_bf16(
          false, a.v, false, b.v, (short)0, acc[nt], false, false);
    }
  }

  // C layout: acc[nt][v] is (row m = v + 8*(lane>=16), col = nt*16 + (lane&15)).
  // Row m's dst id lives in lane m -> lane permute.
  const int mbase = (lane >> 4) << 3;
  int dstm[8]; int okm[8];
  #pragma unroll
  for (int v = 0; v < 8; ++v) {
    dstm[v] = __shfl(dst, mbase + v, 32);
    okm[v] = (e0 + mbase + v) < n_edges;
  }
  const int col = lane & 15;
  #pragma unroll
  for (int nt = 0; nt < NTILE; ++nt) {
    const int f = nt * 16 + col;
    #pragma unroll
    for (int v = 0; v < 8; ++v) {
      if (!okm[v]) continue;
      float val = fmaxf(acc[nt][v], 0.0f);   // relu
      size_t o = (size_t)dstm[v] * DD + f;
      unsafeAtomicAdd(sum + o, val);
      unsafeAtomicAdd(sumsq + o, val * val);
      int bits = __builtin_bit_cast(int, val);   // val>=0: int order == fp order
      atomicMax(mxb + o, bits);
      atomicMin(mnb + o, bits);
    }
  }
  if (lane < 16 && rvalid) unsafeAtomicAdd(deg + dst, 1.0f);
}

// ---- posttrans: [N,1664] x [1664,128]; mean/std finalize + scalers fused ---
__global__ __launch_bounds__(256) void post_kernel(
    const float* __restrict__ h, const float* __restrict__ sumv,
    const float* __restrict__ sqv, const float* __restrict__ mxv,
    const float* __restrict__ mnv, const float* __restrict__ deg,
    const unsigned* __restrict__ wswz, const float* __restrict__ bias,
    float* __restrict__ hp, int n_nodes) {
  const int lane = threadIdx.x & 31;
  const int wt = blockIdx.x * (blockDim.x >> 5) + (threadIdx.x >> 5);
  const int i0 = wt << 4;
  if (i0 >= n_nodes) return;

  const int r = lane & 15;
  const int ir = min(i0 + r, n_nodes - 1);
  const float* hrow   = h    + (size_t)ir * DD;
  const float* sumrow = sumv + (size_t)ir * DD;
  const float* sqrow  = sqv  + (size_t)ir * DD;
  const float* xrow   = mxv  + (size_t)ir * DD;
  const float* nrow   = mnv  + (size_t)ir * DD;
  const float dg = deg[ir];
  const float invd = 1.0f / dg;
  const float ld = logf(dg + 1.0f);
  const float s1 = ld * (1.0f / AVGDLOG);
  const float s2 = AVGDLOG / ld;

  // X[k]: [h | mean,max,min,std | *s1 | *s2], finalization done inline.
  auto xval = [&](int k) -> float {
    if (k < DD) return hrow[k];
    int c = k - DD;
    int si    = c >> 9;       // scaler: 0=id, 1=amp, 2=atten
    int cc    = c & 511;
    int which = cc >> 7;      // 0 mean, 1 max, 2 min, 3 std
    int f     = cc & 127;
    float val;
    if (which == 0) {
      val = sumrow[f] * invd;
    } else if (which == 1) {
      val = xrow[f];
    } else if (which == 2) {
      val = nrow[f];
    } else {
      float mean = sumrow[f] * invd;
      val = sqrtf(fmaxf(sqrow[f] * invd - mean * mean, 0.0f) + EPSV);
    }
    float sc = (si == 0) ? 1.0f : (si == 1) ? s1 : s2;
    return val * sc;
  };

  v8f acc[NTILE];
  #pragma unroll
  for (int nt = 0; nt < NTILE; ++nt) {
    float b = bias[nt * 16 + r];
    #pragma unroll
    for (int j = 0; j < 8; ++j) acc[nt][j] = b;
  }

  for (int ks = 0; ks < 52; ++ks) {       // K = 1664 = 52 * 32
    // stream-prefetch next-next k-step of the weight fragments (global_prefetch)
    if (ks + 2 < 52)
      __builtin_prefetch(wswz + ((size_t)((ks + 2) * NTILE) * 32 + lane) * 8, 0, 0);
    Frag a;
    #pragma unroll
    for (int v = 0; v < 8; ++v) {
      int k0 = ks * 32 + a_koff(v, lane); // even; pair never crosses a 128 block
      a.u[v] = bf16pack(xval(k0), xval(k0 + 1));
    }
    #pragma unroll
    for (int nt = 0; nt < NTILE; ++nt) {
      Frag b;
      load_bfrag(b, wswz, ks, nt, lane);
      acc[nt] = __builtin_amdgcn_wmma_f32_16x16x32_bf16(
          false, a.v, false, b.v, (short)0, acc[nt], false, false);
    }
  }

  const int mbase = (lane >> 4) << 3;
  const int col = lane & 15;
  #pragma unroll
  for (int nt = 0; nt < NTILE; ++nt) {
    #pragma unroll
    for (int v = 0; v < 8; ++v) {
      int im = i0 + mbase + v;
      if (im < n_nodes)
        hp[(size_t)im * DD + nt * 16 + col] = fmaxf(acc[nt][v], 0.0f);
    }
  }
}

// ---- mix: hp x W_mix, leaky relu, residual ---------------------------------
__global__ __launch_bounds__(256) void mix_kernel(
    const float* __restrict__ h, const float* __restrict__ hp,
    const unsigned* __restrict__ wswz, const float* __restrict__ bias,
    float* __restrict__ out, int n_nodes) {
  const int lane = threadIdx.x & 31;
  const int wt = blockIdx.x * (blockDim.x >> 5) + (threadIdx.x >> 5);
  const int i0 = wt << 4;
  if (i0 >= n_nodes) return;

  const int r = lane & 15;
  const int ir = min(i0 + r, n_nodes - 1);
  const float* prow = hp + (size_t)ir * DD;

  v8f acc[NTILE];
  #pragma unroll
  for (int nt = 0; nt < NTILE; ++nt) {
    float b = bias[nt * 16 + r];
    #pragma unroll
    for (int j = 0; j < 8; ++j) acc[nt][j] = b;
  }

  #pragma unroll
  for (int ks = 0; ks < 4; ++ks) {        // K = 128
    Frag a;
    #pragma unroll
    for (int v = 0; v < 8; ++v) {
      const float2 p = *(const float2*)(prow + ks * 32 + a_koff(v, lane));
      a.u[v] = bf16pack(p.x, p.y);
    }
    #pragma unroll
    for (int nt = 0; nt < NTILE; ++nt) {
      Frag b;
      load_bfrag(b, wswz, ks, nt, lane);
      acc[nt] = __builtin_amdgcn_wmma_f32_16x16x32_bf16(
          false, a.v, false, b.v, (short)0, acc[nt], false, false);
    }
  }

  const int mbase = (lane >> 4) << 3;
  const int col = lane & 15;
  #pragma unroll
  for (int nt = 0; nt < NTILE; ++nt) {
    #pragma unroll
    for (int v = 0; v < 8; ++v) {
      int im = i0 + mbase + v;
      if (im < n_nodes) {
        size_t o = (size_t)im * DD + nt * 16 + col;
        float t = acc[nt][v];
        float lr = (t > 0.0f) ? t : 0.01f * t;   // leaky relu
        out[o] = h[o] + lr;                      // residual
      }
    }
  }
}

// ---- host launcher ---------------------------------------------------------
extern "C" void kernel_launch(void* const* d_in, const int* in_sizes, int n_in,
                              void* d_out, int out_size, void* d_ws, size_t ws_size,
                              hipStream_t stream) {
  const float* h      = (const float*)d_in[0];
  const int*   esrc   = (const int*)d_in[1];
  const int*   edst   = (const int*)d_in[2];
  const float* W_pre  = (const float*)d_in[3];
  const float* b_pre  = (const float*)d_in[4];
  const float* W_post = (const float*)d_in[5];
  const float* b_post = (const float*)d_in[6];
  const float* W_mix  = (const float*)d_in[7];
  const float* b_mix  = (const float*)d_in[8];
  const int N = in_sizes[0] / DD;
  const int E = in_sizes[1];

  const int ks_pre = 256 / 32, ks_post = 1664 / 32, ks_mix = 128 / 32;

  char* ws = (char*)d_ws;
  size_t off = 0;
  auto carve = [&](size_t bytes) -> char* {
    char* p = ws + off;
    off += (bytes + 255) & ~(size_t)255;
    return p;
  };
  unsigned* wswz_pre  = (unsigned*)carve((size_t)ks_pre  * 2048 * 4);
  unsigned* wswz_post = (unsigned*)carve((size_t)ks_post * 2048 * 4);
  unsigned* wswz_mix  = (unsigned*)carve((size_t)ks_mix  * 2048 * 4);
  size_t nd_bytes = (size_t)N * DD * sizeof(float);
  float* sumb   = (float*)carve(nd_bytes);
  float* sumsqb = (float*)carve(nd_bytes);
  int*   mxb    = (int*)carve(nd_bytes);
  int*   mnb    = (int*)carve(nd_bytes);
  float* degb   = (float*)carve((size_t)N * sizeof(float));
  float* hpb    = (float*)carve(nd_bytes);

  // 1) swizzle weights into WMMA B-fragment layout
  swz_kernel<<<(ks_pre  * 2048 + 255) / 256, 256, 0, stream>>>(W_pre,  wswz_pre,  ks_pre);
  swz_kernel<<<(ks_post * 2048 + 255) / 256, 256, 0, stream>>>(W_post, wswz_post, ks_post);
  swz_kernel<<<(ks_mix  * 2048 + 255) / 256, 256, 0, stream>>>(W_mix,  wswz_mix,  ks_mix);

  // 2) init aggregation buffers
  int total_nd = N * DD;
  init_kernel<<<(total_nd + 255) / 256, 256, 0, stream>>>(sumb, sumsqb, mxb, mnb, degb, N);

  // 3) edge GEMM + atomic segment reduce (16 edges per wave, 8 waves/block)
  int ewaves = (E + 15) / 16;
  edge_kernel<<<(ewaves + 7) / 8, 256, 0, stream>>>(
      h, esrc, edst, wswz_pre, b_pre, sumb, sumsqb, mxb, mnb, degb, E);

  // 4) posttrans GEMM with fused finalize + scalers
  int nwaves = (N + 15) / 16;
  post_kernel<<<(nwaves + 7) / 8, 256, 0, stream>>>(
      h, sumb, sumsqb, (const float*)mxb, (const float*)mnb, degb,
      wswz_post, b_post, hpb, N);

  // 5) mix GEMM + leaky relu + residual
  mix_kernel<<<(nwaves + 7) / 8, 256, 0, stream>>>(
      h, hpb, wswz_mix, b_mix, (float*)d_out, N);
}
